// AMB3RStage2V4_75737453298213
// MI455X (gfx1250) — matrix operand order repaired and checked by reading, not compiled, not run on hardware
//
#include <hip/hip_runtime.h>
#include <math.h>

// ---------------- types ----------------
typedef __bf16 bf16_t;
typedef __attribute__((ext_vector_type(16))) __bf16 v16bf;
typedef __attribute__((ext_vector_type(8)))  float  v8f;
typedef unsigned int u32x4 __attribute__((ext_vector_type(4)));

#define BTC   16
#define NP    1369
#define NTOT  (BTC*NP)        // 21904
#define CVD   2048
#define MEMD  128
#define HIDD  256
#define NHEAD 4
#define DH    64
#define KST   16384
#define GRID3 (64*64*64)      // 262144

static __device__ __forceinline__ bf16_t f2bf(float f) {
  union { float f; unsigned int u; } a; a.f = f;
  unsigned int r = (a.u + 0x7FFFu + ((a.u >> 16) & 1u)) >> 16;
  union { unsigned short s; bf16_t b; } o; o.s = (unsigned short)r;
  return o.b;
}

union FragU { v16bf v; u32x4 q[2]; unsigned short s[16]; };

// A-operand 16x32 bf16 frag: lanes<16 -> K 0..15; lanes>=16 -> K 16..31
// Caller passes a row already clamped in-bounds; OOB tiles load harmless real data
// (garbage only affects never-stored C rows/cols).
static __device__ __forceinline__ v16bf loadA(const bf16_t* base, long row, int ld, int kb) {
  FragU f;
  int lane = threadIdx.x & 31;
  int k0 = kb + ((lane & 16) ? 16 : 0);
  const bf16_t* p = base + row * (long)ld + k0;
  f.q[0] = *(const u32x4*)(p);
  f.q[1] = *(const u32x4*)(p + 8);
  return f.v;
}
// B-operand 32x16 bf16 frag (col per lane): lanes<16 -> K 0..15; lanes>=16 -> K 16..31
static __device__ __forceinline__ v16bf loadB(const bf16_t* base, long row, int ld, int kb) {
  FragU f;
  int lane = threadIdx.x & 31;
  int k0 = kb + ((lane & 16) ? 16 : 0);
  const bf16_t* p = base + row * (long)ld + k0;
  f.q[0] = *(const u32x4*)(p);
  f.q[1] = *(const u32x4*)(p + 8);
  return f.v;
}

static __device__ __forceinline__ int imin(int a, int b) { return a < b ? a : b; }

// ---------------- generic bf16 WMMA GEMM: C = A(MxK) * B(NnxK)^T + bias ----------------
// modes: 0 f32 out; 1 bf16 out; 2 gelu->f32; 3 fuse: resid+gamma*val -> f32 + bf16; 4 headout
__global__ void gemm_bf16_wmma(const bf16_t* __restrict__ A, const bf16_t* __restrict__ B,
                               int M, int Nn, int K,
                               const float* __restrict__ bias, int mode,
                               float* __restrict__ outF, bf16_t* __restrict__ outB,
                               float* __restrict__ outF2,
                               const float* __restrict__ resid, const float* __restrict__ gammap)
{
  int lane = threadIdx.x & 31;
  int wave = threadIdx.x >> 5;           // 0..3
  int wm = wave & 1, wn = wave >> 1;
  int m0 = blockIdx.y * 64 + wm * 32;
  int n0 = blockIdx.x * 64 + wn * 32;
  int r  = lane & 15;
  // clamped (always in-bounds) source rows; OOB rows never stored
  long mA0 = imin(m0 + r,      M - 1);
  long mA1 = imin(m0 + 16 + r, M - 1);
  long nB0 = imin(n0 + r,      Nn - 1);
  long nB1 = imin(n0 + 16 + r, Nn - 1);
  v8f acc00 = {}, acc01 = {}, acc10 = {}, acc11 = {};
  for (int kb = 0; kb < K; kb += 32) {
    v16bf a0 = loadA(A, mA0, K, kb);
    v16bf a1 = loadA(A, mA1, K, kb);
    v16bf b0 = loadB(B, nB0, K, kb);
    v16bf b1 = loadB(B, nB1, K, kb);
    acc00 = __builtin_amdgcn_wmma_f32_16x16x32_bf16(false, a0, false, b0, (short)0, acc00, false, false);
    acc01 = __builtin_amdgcn_wmma_f32_16x16x32_bf16(false, a0, false, b1, (short)0, acc01, false, false);
    acc10 = __builtin_amdgcn_wmma_f32_16x16x32_bf16(false, a1, false, b0, (short)0, acc10, false, false);
    acc11 = __builtin_amdgcn_wmma_f32_16x16x32_bf16(false, a1, false, b1, (short)0, acc11, false, false);
  }
  float gamma = gammap ? gammap[0] : 0.f;
  int rofs = (lane & 16) ? 8 : 0;
  for (int i = 0; i < 2; i++) {
    for (int j = 0; j < 2; j++) {
      v8f acc = (i==0) ? (j==0?acc00:acc01) : (j==0?acc10:acc11);
      for (int e = 0; e < 8; e++) {
        int m = m0 + i*16 + e + rofs;
        int n = n0 + j*16 + r;
        if (m < M && n < Nn) {
          float val = acc[e];
          if (bias) val += bias[n];
          long idx = (long)m * Nn + n;
          if (mode == 0) {
            outF[idx] = val;
          } else if (mode == 1) {
            outB[idx] = f2bf(val);
          } else if (mode == 2) {
            outF[idx] = 0.5f * val * (1.0f + erff(val * 0.70710678118654752f));
          } else if (mode == 3) {
            float xf = resid[idx] + gamma * val;
            outF[idx] = xf; outB[idx] = f2bf(xf);
          } else { // 4: mlp2 head split
            if (n < MEMD)       outF[(long)m*MEMD + n] = val;
            else if (n == MEMD) outF2[m] = 1.0f / (1.0f + expf(-val));
          }
        }
      }
    }
  }
}

// ---------------- flash attention (one wave per (b, head, 16-row q tile)) ----------------
__global__ void attn_flash(const bf16_t* __restrict__ Q, const bf16_t* __restrict__ Km,
                           const bf16_t* __restrict__ Vm, bf16_t* __restrict__ AO)
{
  __shared__ bf16_t Plds[16 * 48];   // P tile 16x32, stride 48 (96B rows)
  __shared__ bf16_t Vlds[32 * 72];   // V chunk 32x64, stride 72 (144B rows)
  int lane = threadIdx.x;
  int qt = blockIdx.x, h = blockIdx.y, b = blockIdx.z;
  const bf16_t* qb = Q  + ((long)b * NP) * HIDD + h * DH;
  const bf16_t* kp = Km + ((long)b * NP) * HIDD + h * DH;
  const bf16_t* vp = Vm + ((long)b * NP) * HIDD + h * DH;
  int r = lane & 15;
  int rofs = (lane & 16) ? 8 : 0;
  int qrow = qt * 16 + r;
  v16bf aq0 = loadA(qb, imin(qrow, NP - 1), HIDD, 0);
  v16bf aq1 = loadA(qb, imin(qrow, NP - 1), HIDD, 32);
  v8f o0 = {}, o1 = {}, o2 = {}, o3 = {};
  float ms[8], ls[8];
  for (int e = 0; e < 8; e++) { ms[e] = -1e30f; ls[e] = 0.f; }
  const float scale = 0.125f;  // 1/sqrt(64)

  for (int kc = 0; kc < NP; kc += 32) {
    int key0 = kc + r, key1 = kc + 16 + r;
    long kr0 = imin(key0, NP - 1), kr1 = imin(key1, NP - 1);
    v8f s0 = {}, s1 = {};
    {
      v16bf bk00 = loadB(kp, kr0, HIDD, 0);
      v16bf bk01 = loadB(kp, kr0, HIDD, 32);
      v16bf bk10 = loadB(kp, kr1, HIDD, 0);
      v16bf bk11 = loadB(kp, kr1, HIDD, 32);
      s0 = __builtin_amdgcn_wmma_f32_16x16x32_bf16(false, aq0, false, bk00, (short)0, s0, false, false);
      s0 = __builtin_amdgcn_wmma_f32_16x16x32_bf16(false, aq1, false, bk01, (short)0, s0, false, false);
      s1 = __builtin_amdgcn_wmma_f32_16x16x32_bf16(false, aq0, false, bk10, (short)0, s1, false, false);
      s1 = __builtin_amdgcn_wmma_f32_16x16x32_bf16(false, aq1, false, bk11, (short)0, s1, false, false);
    }
    // stage V chunk: 32 bf16 (4x16B) of one key row per lane (zero-fill OOB keys)
    {
      int key = kc + lane;
      u32x4* dst = (u32x4*)(&Vlds[lane * 72]);
      if (key < NP) {
        const u32x4* src = (const u32x4*)(vp + (long)key * HIDD);
        dst[0] = src[0]; dst[1] = src[1]; dst[2] = src[2]; dst[3] = src[3];
      } else {
        u32x4 z = {0u,0u,0u,0u};
        dst[0] = z; dst[1] = z; dst[2] = z; dst[3] = z;
      }
    }
    // online softmax (rows live in half-waves; width-16 shuffles reduce over cols)
    float p0[8], p1[8], corr[8];
    for (int e = 0; e < 8; e++) {
      float v0 = (key0 < NP) ? s0[e] * scale : -1e30f;
      float v1 = (key1 < NP) ? s1[e] * scale : -1e30f;
      float t = fmaxf(v0, v1);
      for (int off = 8; off >= 1; off >>= 1) t = fmaxf(t, __shfl_xor(t, off, 16));
      float nm = fmaxf(ms[e], t);
      float c  = expf(ms[e] - nm);
      float e0 = expf(v0 - nm), e1 = expf(v1 - nm);
      float rs = e0 + e1;
      for (int off = 8; off >= 1; off >>= 1) rs += __shfl_xor(rs, off, 16);
      ls[e] = ls[e] * c + rs;
      ms[e] = nm; corr[e] = c; p0[e] = e0; p1[e] = e1;
    }
    for (int e = 0; e < 8; e++) { o0[e]*=corr[e]; o1[e]*=corr[e]; o2[e]*=corr[e]; o3[e]*=corr[e]; }
    __syncthreads();
    for (int e = 0; e < 8; e++) {
      int row = e + rofs;
      Plds[row*48 + r]      = f2bf(p0[e]);
      Plds[row*48 + 16 + r] = f2bf(p1[e]);
    }
    __syncthreads();
    // P as A-frag from LDS
    v16bf aP;
    {
      FragU f;
      const bf16_t* p = &Plds[r*48 + ((lane & 16) ? 16 : 0)];
      f.q[0] = *(const u32x4*)(p);
      f.q[1] = *(const u32x4*)(p + 8);
      aP = f.v;
    }
    // V as B-frags gathered from LDS; accumulate P·V
    int kb0 = (lane & 16) ? 16 : 0;
    const unsigned short* vl = (const unsigned short*)Vlds;
    for (int d = 0; d < 4; d++) {
      FragU f;
      for (int t = 0; t < 16; t++) f.s[t] = vl[(kb0 + t)*72 + d*16 + r];
      if      (d == 0) o0 = __builtin_amdgcn_wmma_f32_16x16x32_bf16(false, aP, false, f.v, (short)0, o0, false, false);
      else if (d == 1) o1 = __builtin_amdgcn_wmma_f32_16x16x32_bf16(false, aP, false, f.v, (short)0, o1, false, false);
      else if (d == 2) o2 = __builtin_amdgcn_wmma_f32_16x16x32_bf16(false, aP, false, f.v, (short)0, o2, false, false);
      else             o3 = __builtin_amdgcn_wmma_f32_16x16x32_bf16(false, aP, false, f.v, (short)0, o3, false, false);
    }
    __syncthreads();
  }
  for (int e = 0; e < 8; e++) {
    int row = qt*16 + e + rofs;
    if (row < NP) {
      float inv = 1.0f / ls[e];
      bf16_t* dst = AO + ((long)b*NP + row)*HIDD + h*DH;
      dst[0*16 + r] = f2bf(o0[e]*inv);
      dst[1*16 + r] = f2bf(o1[e]*inv);
      dst[2*16 + r] = f2bf(o2[e]*inv);
      dst[3*16 + r] = f2bf(o3[e]*inv);
    }
  }
}

// ---------------- small helper kernels ----------------
__global__ void cvt_bf16(const float* __restrict__ in, bf16_t* __restrict__ out, long n) {
  long i = (long)blockIdx.x*blockDim.x + threadIdx.x;
  long stride = (long)gridDim.x*blockDim.x;
  for (; i < n; i += stride) out[i] = f2bf(in[i]);
}

// C[o,i] = sum_j A[o,j]*B[j,i]  (A row-stride J, B row-stride I) -> bf16
__global__ void wcomb(const float* __restrict__ A, const float* __restrict__ B,
                      bf16_t* __restrict__ C, int O, int J, int I) {
  int idx = blockIdx.x*blockDim.x + threadIdx.x;
  if (idx >= O*I) return;
  int o = idx / I, i = idx % I;
  float s = 0.f;
  for (int j = 0; j < J; j++) s += A[(long)o*J + j] * B[(long)j*I + i];
  C[(long)o*I + i] = f2bf(s);
}

__global__ void biascomb(const float* __restrict__ W, const float* __restrict__ bvec,
                         float* __restrict__ out, int O, int J) {
  int o = blockIdx.x*blockDim.x + threadIdx.x;
  if (o >= O) return;
  float s = 0.f;
  for (int j = 0; j < J; j++) s += W[(long)o*J + j] * bvec[j];
  out[o] = s;
}

__global__ void hash_gather(const float* __restrict__ xyz, const int* __restrict__ sh,
                            const float* __restrict__ sv, const float* __restrict__ nulltok,
                            bf16_t* __restrict__ xmem, int* __restrict__ hout,
                            float* __restrict__ mask_out) {
  int i = blockIdx.x, c = threadIdx.x;
  int vx = (int)floorf(xyz[i*3+0] * 20.0f);
  int vy = (int)floorf(xyz[i*3+1] * 20.0f);
  int vz = (int)floorf(xyz[i*3+2] * 20.0f);
  int h = (vx * 64 + vy) * 64 + vz;
  int lo = 0, hi = KST;
  while (lo < hi) { int mid = (lo + hi) >> 1; if (sh[mid] < h) lo = mid + 1; else hi = mid; }
  int pos = lo > KST-1 ? KST-1 : lo;
  bool hit = (sh[pos] == h);
  float f = hit ? sv[(long)pos*MEMD + c] : nulltok[c];
  xmem[(long)i*MEMD + c] = f2bf(f);
  if (c == 0) { hout[i] = h; mask_out[i] = hit ? 1.0f : 0.0f; }
}

__global__ void layernorm_k(const float* __restrict__ h1, const float* __restrict__ g,
                            const float* __restrict__ bvec, bf16_t* __restrict__ out, int M) {
  int wave = threadIdx.x >> 5, lane = threadIdx.x & 31;
  int row = blockIdx.x * 8 + wave;
  if (row >= M) return;
  const float* x = h1 + (long)row * 512;
  float vals[16], s = 0.f, sq = 0.f;
  for (int t = 0; t < 16; t++) { float v = x[lane + 32*t]; vals[t] = v; s += v; sq += v*v; }
  for (int off = 16; off >= 1; off >>= 1) { s += __shfl_xor(s, off, 32); sq += __shfl_xor(sq, off, 32); }
  float mean = s * (1.f/512.f);
  float var  = sq * (1.f/512.f) - mean*mean;
  float rstd = rsqrtf(var + 1e-5f);
  bf16_t* o = out + (long)row * 512;
  for (int t = 0; t < 16; t++) {
    int c = lane + 32*t;
    o[c] = f2bf((vals[t] - mean) * rstd * g[c] + bvec[c]);
  }
}

__global__ void set_flags(const int* __restrict__ h, int* __restrict__ flags, int n) {
  int i = blockIdx.x*blockDim.x + threadIdx.x;
  if (i < n) flags[h[i]] = 1;
}

__global__ void scan_flags(const int* __restrict__ flags, int* __restrict__ rank,
                           int* __restrict__ uhash, int* __restrict__ numU) {
  __shared__ int sh[1024];
  int tid = threadIdx.x;
  int running = 0;
  for (int chunk = 0; chunk < GRID3; chunk += 1024) {
    int idx = chunk + tid;
    int f = flags[idx];
    sh[tid] = f;
    __syncthreads();
    for (int off = 1; off < 1024; off <<= 1) {
      int t = (tid >= off) ? sh[tid - off] : 0;
      __syncthreads();
      sh[tid] += t;
      __syncthreads();
    }
    int incl = sh[tid];
    int r = running + incl - f;
    rank[idx] = r;
    if (f) uhash[r] = idx;
    running += sh[1023];
    __syncthreads();
  }
  if (tid == 0) numU[0] = running;
}

__global__ void accum_vox(const int* __restrict__ h, const int* __restrict__ rank,
                          const float* __restrict__ mread, const float* __restrict__ wconf,
                          float* __restrict__ num, float* __restrict__ den) {
  int i = blockIdx.x, c = threadIdx.x;
  int r = rank[h[i]];
  float w = wconf[i];
  atomicAdd(&num[(long)r*MEMD + c], w * mread[(long)i*MEMD + c]);
  if (c == 0) atomicAdd(&den[r], w);
}

__global__ void finalize_vox(const int* __restrict__ uhash, const int* __restrict__ numU,
                             const int* __restrict__ sh, const float* __restrict__ sv,
                             const float* __restrict__ num, const float* __restrict__ den,
                             float* __restrict__ out) {
  int rr = blockIdx.x, c = threadIdx.x;
  float val = 0.f;
  if (rr < numU[0]) {
    float vn = num[(long)rr*MEMD + c] / (den[rr] + 1e-8f);
    int hv = uhash[rr];
    int lo = 0, hi = KST;
    while (lo < hi) { int mid = (lo + hi) >> 1; if (sh[mid] < hv) lo = mid + 1; else hi = mid; }
    int pos = lo > KST-1 ? KST-1 : lo;
    val = (sh[pos] == hv) ? (0.5f * sv[(long)pos*MEMD + c] + 0.5f * vn) : vn;
  }
  out[(long)rr*MEMD + c] = val;
}

// ---------------- host launcher ----------------
extern "C" void kernel_launch(void* const* d_in, const int* in_sizes, int n_in,
                              void* d_out, int out_size, void* d_ws, size_t ws_size,
                              hipStream_t stream) {
  const float* X_vggt     = (const float*)d_in[0];
  const float* xyz        = (const float*)d_in[1];
  const int*   store_hash = (const int*)  d_in[2];
  const float* store_vals = (const float*)d_in[3];
  const float* null_token = (const float*)d_in[4];
  const float* Wq_f       = (const float*)d_in[5];
  const float* Wk_f       = (const float*)d_in[6];
  const float* Wv_f       = (const float*)d_in[7];
  const float* in_proj_w  = (const float*)d_in[8];
  const float* in_proj_b  = (const float*)d_in[9];
  const float* attn_out_w = (const float*)d_in[10];
  const float* attn_out_b = (const float*)d_in[11];
  const float* fuse_out_w = (const float*)d_in[12];
  const float* gammap     = (const float*)d_in[13];
  const float* mlp1_w     = (const float*)d_in[14];
  const float* mlp1_b     = (const float*)d_in[15];
  const float* ln_g       = (const float*)d_in[16];
  const float* ln_b       = (const float*)d_in[17];
  const float* mlp2_w     = (const float*)d_in[18];
  const float* mlp2_b     = (const float*)d_in[19];
  (void)in_sizes; (void)n_in; (void)out_size; (void)ws_size;

  float* out = (float*)d_out;
  const size_t XFUSE_OFF = 0;
  const size_t MREAD_OFF = (size_t)NTOT * CVD;                 // 44,859,392
  const size_t WCONF_OFF = MREAD_OFF + (size_t)NTOT * MEMD;    // 47,663,104
  const size_t VOX_OFF   = WCONF_OFF + (size_t)NTOT;           // 47,685,008
  const size_t MASK_OFF  = VOX_OFF + (size_t)NTOT * MEMD;      // 50,488,720

  char* ws = (char*)d_ws;
  size_t off = 0;
  auto alloc = [&](size_t bytes) -> void* {
    void* p = ws + off;
    off = (off + bytes + 255) & ~(size_t)255;
    return p;
  };
  bf16_t* xv_bf    = (bf16_t*)alloc((size_t)NTOT * CVD * 2);   // later reused as X_fuse bf16
  bf16_t* xmem_bf  = (bf16_t*)alloc((size_t)NTOT * MEMD * 2);
  int*    h_i32    = (int*)   alloc((size_t)NTOT * 4);
  bf16_t* wq_eff   = (bf16_t*)alloc((size_t)HIDD * CVD * 2);
  bf16_t* wk_eff   = (bf16_t*)alloc((size_t)HIDD * MEMD * 2);
  bf16_t* wv_eff   = (bf16_t*)alloc((size_t)HIDD * MEMD * 2);
  bf16_t* w2_eff   = (bf16_t*)alloc((size_t)CVD * HIDD * 2);
  float*  bias2    = (float*) alloc((size_t)CVD * 4);
  bf16_t* mlp1w_bf = (bf16_t*)alloc((size_t)512 * CVD * 2);
  bf16_t* mlp2w_bf = (bf16_t*)alloc((size_t)129 * 512 * 2);
  bf16_t* q_bf     = (bf16_t*)alloc((size_t)NTOT * HIDD * 2);
  bf16_t* k_bf     = (bf16_t*)alloc((size_t)NTOT * HIDD * 2);
  bf16_t* v_bf     = (bf16_t*)alloc((size_t)NTOT * HIDD * 2);
  bf16_t* ao_bf    = (bf16_t*)alloc((size_t)NTOT * HIDD * 2);
  float*  h1_f     = (float*) alloc((size_t)NTOT * 512 * 4);
  bf16_t* h1n_bf   = (bf16_t*)alloc((size_t)NTOT * 512 * 2);
  int*    flags    = (int*)   alloc((size_t)GRID3 * 4);
  int*    rank     = (int*)   alloc((size_t)GRID3 * 4);
  int*    uhash    = (int*)   alloc((size_t)NTOT * 4);
  int*    numU     = (int*)   alloc(256);
  float*  num_f    = (float*) alloc((size_t)NTOT * MEMD * 4);
  float*  den_f    = (float*) alloc((size_t)NTOT * 4);
  bf16_t* xfuse_bf = xv_bf;  // reuse

  hipMemsetAsync(flags, 0, (size_t)GRID3 * 4, stream);
  hipMemsetAsync(num_f, 0, (size_t)NTOT * MEMD * 4, stream);
  hipMemsetAsync(den_f, 0, (size_t)NTOT * 4, stream);

  // convert inputs to bf16
  cvt_bf16<<<4096, 256, 0, stream>>>(X_vggt, xv_bf, (long)NTOT * CVD);
  cvt_bf16<<<1024, 256, 0, stream>>>(mlp1_w, mlp1w_bf, (long)512 * CVD);
  cvt_bf16<<<256, 256, 0, stream>>>(mlp2_w, mlp2w_bf, (long)129 * 512);

  // combined weights
  wcomb<<<(HIDD*CVD + 255)/256, 256, 0, stream>>>(in_proj_w,               Wq_f,       wq_eff, HIDD, HIDD, CVD);
  wcomb<<<(HIDD*MEMD + 255)/256, 256, 0, stream>>>(in_proj_w + HIDD*HIDD,   Wk_f,       wk_eff, HIDD, HIDD, MEMD);
  wcomb<<<(HIDD*MEMD + 255)/256, 256, 0, stream>>>(in_proj_w + 2*HIDD*HIDD, Wv_f,       wv_eff, HIDD, HIDD, MEMD);
  wcomb<<<(CVD*HIDD + 255)/256, 256, 0, stream>>>(fuse_out_w,              attn_out_w, w2_eff, CVD,  HIDD, HIDD);
  biascomb<<<(CVD + 255)/256, 256, 0, stream>>>(fuse_out_w, attn_out_b, bias2, CVD, HIDD);

  // hash / gather memory tokens (+ mask output)
  hash_gather<<<NTOT, 128, 0, stream>>>(xyz, store_hash, store_vals, null_token,
                                        xmem_bf, h_i32, out + MASK_OFF);

  const int MB = (NTOT + 63) / 64; // 343
  // q / k / v projections (fused input proj)
  gemm_bf16_wmma<<<dim3(HIDD/64, MB), 128, 0, stream>>>(xv_bf, wq_eff, NTOT, HIDD, CVD,
      in_proj_b, 1, nullptr, q_bf, nullptr, nullptr, nullptr);
  gemm_bf16_wmma<<<dim3(HIDD/64, MB), 128, 0, stream>>>(xmem_bf, wk_eff, NTOT, HIDD, MEMD,
      in_proj_b + HIDD, 1, nullptr, k_bf, nullptr, nullptr, nullptr);
  gemm_bf16_wmma<<<dim3(HIDD/64, MB), 128, 0, stream>>>(xmem_bf, wv_eff, NTOT, HIDD, MEMD,
      in_proj_b + 2*HIDD, 1, nullptr, v_bf, nullptr, nullptr, nullptr);

  // flash attention
  attn_flash<<<dim3((NP + 15)/16, NHEAD, BTC), 32, 0, stream>>>(q_bf, k_bf, v_bf, ao_bf);

  // delta = ao @ (fuse_out_w@attn_out_w)^T + fuse_out_w@attn_out_b ; X_fuse = X + gamma*delta
  gemm_bf16_wmma<<<dim3(CVD/64, MB), 128, 0, stream>>>(ao_bf, w2_eff, NTOT, CVD, HIDD,
      bias2, 3, out + XFUSE_OFF, xfuse_bf, nullptr, X_vggt, gammap);

  // mlp1 + exact gelu
  gemm_bf16_wmma<<<dim3(512/64, MB), 128, 0, stream>>>(xfuse_bf, mlp1w_bf, NTOT, 512, CVD,
      mlp1_b, 2, h1_f, nullptr, nullptr, nullptr, nullptr);

  layernorm_k<<<(NTOT + 7)/8, 256, 0, stream>>>(h1_f, ln_g, ln_b, h1n_bf, NTOT);

  // mlp2 -> M_readout + sigmoid W_conf
  gemm_bf16_wmma<<<dim3((129 + 63)/64, MB), 128, 0, stream>>>(h1n_bf, mlp2w_bf, NTOT, 129, 512,
      mlp2_b, 4, out + MREAD_OFF, nullptr, out + WCONF_OFF, nullptr, nullptr);

  // voxel pooling: unique(sorted) via dense flag table + scan, segment mean, alpha blend
  set_flags<<<(NTOT + 255)/256, 256, 0, stream>>>(h_i32, flags, NTOT);
  scan_flags<<<1, 1024, 0, stream>>>(flags, rank, uhash, numU);
  accum_vox<<<NTOT, 128, 0, stream>>>(h_i32, rank, out + MREAD_OFF, out + WCONF_OFF, num_f, den_f);
  finalize_vox<<<NTOT, 128, 0, stream>>>(uhash, numU, store_hash, store_vals, num_f, den_f,
                                         out + VOX_OFF);
}
